// MultiheadAttention_30064771072843
// MI455X (gfx1250) — compile-verified, matrix operand order checked by reading, not run on hardware
//
#include <hip/hip_runtime.h>
#include <hip/hip_bf16.h>

typedef __bf16 bf16;
typedef __attribute__((ext_vector_type(16))) __bf16 v16bf;
typedef __attribute__((ext_vector_type(8)))  __bf16 v8bf;
typedef __attribute__((ext_vector_type(8)))  float  v8f;

#define WMMA_BF16(a, b, c) \
  __builtin_amdgcn_wmma_f32_16x16x32_bf16(false, (a), false, (b), (short)0, (c), false, false)

#if defined(__has_builtin)
#if __has_builtin(__builtin_amdgcn_tensor_load_to_lds)
#define HAS_TDM 1
#endif
#if __has_builtin(__builtin_amdgcn_mov_dpp8)
#define HAS_DPP8 1
#endif
#endif
#ifndef HAS_TDM
#define HAS_TDM 0
#endif
#ifndef HAS_DPP8
#define HAS_DPP8 0
#endif

// --------------------------------------------------------------------------
// 16-bit WMMA fragment load (K-dim 32) from k-contiguous storage.
// lane l -> row/col (l&15), kbase=(l>>4)*8; elems 0..7 at K=kbase.., 8..15 at K=16+kbase..
// --------------------------------------------------------------------------
__device__ __forceinline__ v16bf frag_ld(const bf16* p) {
  v8bf lo = *(const v8bf*)(p);
  v8bf hi = *(const v8bf*)(p + 16);
  v16bf r;
#pragma unroll
  for (int i = 0; i < 8; ++i) { r[i] = lo[i]; r[i + 8] = hi[i]; }
  return r;
}

__device__ __forceinline__ uint2 pk4(float x, float y, float z, float w) {
  union { bf16 h[4]; uint2 u; } p;
  p.h[0] = (bf16)x; p.h[1] = (bf16)y; p.h[2] = (bf16)z; p.h[3] = (bf16)w;
  return p.u;
}

// --------------------------- cross-lane reductions -------------------------
#if HAS_DPP8
template <int A0, int A1, int A2, int A3, int A4, int A5, int A6, int A7>
__device__ __forceinline__ float dpp8f(float x) {
  constexpr int sel =
      A0 | (A1 << 3) | (A2 << 6) | (A3 << 9) | (A4 << 12) | (A5 << 15) | (A6 << 18) | (A7 << 21);
  return __builtin_bit_cast(float,
      __builtin_amdgcn_mov_dpp8(__builtin_bit_cast(int, x), sel));
}
__device__ __forceinline__ float red16_max(float x) {
  x = fmaxf(x, dpp8f<1, 0, 3, 2, 5, 4, 7, 6>(x));
  x = fmaxf(x, dpp8f<2, 3, 0, 1, 6, 7, 4, 5>(x));
  x = fmaxf(x, dpp8f<4, 5, 6, 7, 0, 1, 2, 3>(x));
  return fmaxf(x, __shfl_xor(x, 8, 32));
}
__device__ __forceinline__ float red16_sum(float x) {
  x += dpp8f<1, 0, 3, 2, 5, 4, 7, 6>(x);
  x += dpp8f<2, 3, 0, 1, 6, 7, 4, 5>(x);
  x += dpp8f<4, 5, 6, 7, 0, 1, 2, 3>(x);
  return x + __shfl_xor(x, 8, 32);
}
#else
__device__ __forceinline__ float red16_max(float x) {
  for (int off = 1; off < 16; off <<= 1) x = fmaxf(x, __shfl_xor(x, off, 32));
  return x;
}
__device__ __forceinline__ float red16_sum(float x) {
  for (int off = 1; off < 16; off <<= 1) x += __shfl_xor(x, off, 32);
  return x;
}
#endif

// ------------------------------- TDM helper --------------------------------
#if HAS_TDM
typedef unsigned int u32x4 __attribute__((ext_vector_type(4)));
typedef int i32x4 __attribute__((ext_vector_type(4)));
typedef int i32x8 __attribute__((ext_vector_type(8)));

__device__ __forceinline__ void tdm_load_2d(unsigned lds_off, const void* gptr,
                                            unsigned rows, unsigned rowlen_elems,
                                            unsigned row_stride_elems, unsigned pad_en,
                                            unsigned pad_interval_code,
                                            unsigned pad_amount_code) {
  const unsigned long long ga = (unsigned long long)(uintptr_t)gptr;
  u32x4 g0;
  g0[0] = 1u;                                                // count=1 (valid user D#)
  g0[1] = lds_off;                                           // lds_addr
  g0[2] = (unsigned)ga;                                      // global_addr[31:0]
  g0[3] = ((unsigned)(ga >> 32) & 0x01FFFFFFu) | (2u << 30); // global_addr[56:32]|type=2
  i32x8 g1;
  g1[0] = (int)((1u << 16) | (pad_en << 20) |                // data_size=2B, pad ctl
                (pad_interval_code << 22) | (pad_amount_code << 25));
  g1[1] = (int)(rowlen_elems << 16);                         // tensor_dim0[15:0]
  g1[2] = (int)(rows << 16);                                 // tensor_dim1[15:0]
  g1[3] = (int)(rowlen_elems << 16);                         // tile_dim0
  g1[4] = (int)rows;                                         // tile_dim1 (tile_dim2=0)
  g1[5] = (int)row_stride_elems;                             // tensor_dim0_stride[31:0]
  g1[6] = 0;
  g1[7] = 0;
  i32x4 z4 = {};
#if __has_include(<hip/amd_detail/amd_gfx1250_TDM.h>)
  i32x8 z8 = {};
  __builtin_amdgcn_tensor_load_to_lds(g0, g1, z4, z4, z8, 0);
#else
  __builtin_amdgcn_tensor_load_to_lds(g0, g1, z4, z4, 0);
#endif
}
#endif

// ---------------------------------------------------------------------------
// Tiled GEMM: C[M x NC] = A[M x K] * B[K x NC] (+bias). Double-buffered LDS,
// 256x128 block tile, 8 waves each computing 64x64 (16 WMMAs : 16 b128 loads
// per K-step -> 1.0 LDS loads per WMMA). K-step 32, bf16 WMMA / fp32 acc.
// LDS row stride 40 elems (80 B): conflict-free b128 frags, matches TDM pad.
// SCALE_Q: multiply cols < 1024 by dim^-0.5 (folds attention scale into qkv).
// ---------------------------------------------------------------------------
template <int NC, bool A_BF16, bool BIAS, bool OUT_BF16, bool SCALE_Q>
__global__ __launch_bounds__(256) void gemm128(const void* __restrict__ Ap,
                                               const float* __restrict__ Bp,
                                               const float* __restrict__ bias,
                                               void* __restrict__ Cp, int K) {
  constexpr int LDA = 40;
  constexpr bool A_TDM = A_BF16 && (HAS_TDM != 0);
  __shared__ __align__(16) bf16 As[2][256 * LDA];  // 40 KB
  __shared__ __align__(16) bf16 Bs[2][128 * LDA];  // 20 KB, transposed: Bs[n][k]

  const int t = threadIdx.x;
  const int bm = blockIdx.y * 256, bn = blockIdx.x * 128;
  const int w = t >> 5, lane = t & 31;
  const int wr = (w >> 1) * 64, wc = (w & 1) * 64;  // 4 waves along M, 2 along N
  const int lh = lane & 15, hh = lane >> 4, kb = hh * 8;

  const int c4 = (t & 7) * 4, ra = t >> 3;   // A staging: rows ra+32p (p<8), cols c4..+3
  const int nc = (t & 31) * 4, rb = t >> 5;  // B staging: rows rb+8p  (p<4), cols nc..+3

  float4 afr[8];
  uint2 ahr[8];
  float4 bst[4];

  auto fetch = [&](int kk, int buf) {
    if constexpr (A_TDM) {
#if HAS_TDM
      if (w == 0)
        tdm_load_2d((unsigned)(uintptr_t)&As[buf][0],
                    (const bf16*)Ap + (size_t)bm * K + kk, 256, 32, (unsigned)K,
                    /*pad*/ 1, /*16 DW*/ 3, /*4 DW*/ 3);
#endif
    } else {
#pragma unroll
      for (int p = 0; p < 8; ++p) {
        if constexpr (A_BF16)
          ahr[p] = *(const uint2*)((const bf16*)Ap + (size_t)(bm + ra + p * 32) * K + kk + c4);
        else
          afr[p] = *(const float4*)((const float*)Ap + (size_t)(bm + ra + p * 32) * K + kk + c4);
      }
    }
#pragma unroll
    for (int p = 0; p < 4; ++p)
      bst[p] = *(const float4*)(Bp + (size_t)(kk + rb + p * 8) * NC + bn + nc);
    if (kk + 32 < K)  // speculative prefetch of the tile after next
      __builtin_prefetch(Bp + (size_t)(kk + 32 + rb) * NC + bn + nc, 0, 0);
  };

  auto stage = [&](int buf) {
    if constexpr (A_TDM) {
#if HAS_TDM
      if (w == 0) __builtin_amdgcn_s_wait_tensorcnt((short)0);
#endif
    } else {
#pragma unroll
      for (int p = 0; p < 8; ++p) {
        const int r = ra + p * 32;
        if constexpr (A_BF16)
          *(uint2*)&As[buf][r * LDA + c4] = ahr[p];
        else
          *(uint2*)&As[buf][r * LDA + c4] = pk4(afr[p].x, afr[p].y, afr[p].z, afr[p].w);
      }
    }
#pragma unroll
    for (int p = 0; p < 4; ++p) {
      const int k = rb + p * 8;
      Bs[buf][(nc + 0) * LDA + k] = (bf16)bst[p].x;
      Bs[buf][(nc + 1) * LDA + k] = (bf16)bst[p].y;
      Bs[buf][(nc + 2) * LDA + k] = (bf16)bst[p].z;
      Bs[buf][(nc + 3) * LDA + k] = (bf16)bst[p].w;
    }
  };

  v8f acc[4][4];
#pragma unroll
  for (int i = 0; i < 4; ++i)
#pragma unroll
    for (int j = 0; j < 4; ++j) acc[i][j] = {};

  fetch(0, 0);
  stage(0);
  __syncthreads();

  int cur = 0;
  for (int kk = 0; kk < K; kk += 32) {
    const int nxt = cur ^ 1;
    if (kk + 32 < K) fetch(kk + 32, nxt);

    v16bf af[4], bfr[4];
#pragma unroll
    for (int mi = 0; mi < 4; ++mi)
      af[mi] = frag_ld(&As[cur][(wr + mi * 16 + lh) * LDA + kb]);
#pragma unroll
    for (int ni = 0; ni < 4; ++ni)
      bfr[ni] = frag_ld(&Bs[cur][(wc + ni * 16 + lh) * LDA + kb]);
#pragma unroll
    for (int mi = 0; mi < 4; ++mi)
#pragma unroll
      for (int ni = 0; ni < 4; ++ni)
        acc[mi][ni] = WMMA_BF16(af[mi], bfr[ni], acc[mi][ni]);

    if (kk + 32 < K) stage(nxt);
    __syncthreads();
    cur = nxt;
  }

  // epilogue: C layout = lane(col); VGPR r -> row r (lanes 0-15) / 8+r (16-31)
#pragma unroll
  for (int mi = 0; mi < 4; ++mi) {
#pragma unroll
    for (int ni = 0; ni < 4; ++ni) {
      const int col = bn + wc + ni * 16 + lh;
      const float bv = BIAS ? bias[col] : 0.0f;
      const float sc = (SCALE_Q && col < 1024) ? 0.03125f : 1.0f;  // dim^-0.5
#pragma unroll
      for (int r = 0; r < 8; ++r) {
        const int row = bm + wr + mi * 16 + hh * 8 + r;
        const float val = (acc[mi][ni][r] + bv) * sc;
        if constexpr (OUT_BF16)
          ((bf16*)Cp)[(size_t)row * NC + col] = (bf16)val;
        else
          ((float*)Cp)[(size_t)row * NC + col] = val;
      }
    }
  }
}

// ---------------------------------------------------------------------------
// Flash attention: block = (128 q-rows, head, batch); 8 waves x 16 rows.
// Double-buffered K/V tiles; K tile staged by TDM (wave 0) when available.
// qkv: [b*1024+n][3072] bf16 (Q*scale | K | V); Q already scaled by GEMM1.
// ---------------------------------------------------------------------------
__global__ __launch_bounds__(256) void attn_fa(const bf16* __restrict__ qkv,
                                               bf16* __restrict__ att) {
  constexpr int LDK = 72, LDV = 40, LDP = 40;  // conflict-free padded strides
  __shared__ __align__(16) bf16 Ks[2][32 * LDK];
  __shared__ __align__(16) bf16 Vs[2][64 * LDV];
  __shared__ __align__(16) bf16 Ps[8 * 16 * LDP];

  const int t = threadIdx.x;
  const int w = t >> 5, lane = t & 31;
  const int lh = lane & 15, hh = lane >> 4, kb = hh * 8;
  const int b = blockIdx.z, h = blockIdx.y;
  const int qr0 = blockIdx.x * 128 + w * 16;
  const size_t rs = 3072;

  // Q fragments stay in registers the whole loop
  const bf16* qrow = qkv + ((size_t)b * 1024 + qr0 + lh) * rs + h * 64;
  v16bf qf[2];
#pragma unroll
  for (int c = 0; c < 2; ++c) qf[c] = frag_ld(qrow + c * 32 + kb);

  const int rv = t >> 3, c8 = (t & 7) * 8;
  v8bf kreg, vreg;

  auto fetch = [&](int kt, int buf) {
    const size_t rowb = ((size_t)b * 1024 + kt * 32 + rv) * rs + h * 64 + c8;
#if HAS_TDM
    (void)rowb;
    if (w == 0)
      tdm_load_2d((unsigned)(uintptr_t)&Ks[buf][0],
                  qkv + ((size_t)b * 1024 + kt * 32) * rs + 1024 + h * 64, 32, 64, 3072,
                  /*pad*/ 1, /*32 DW*/ 4, /*4 DW*/ 3);
    vreg = *(const v8bf*)(qkv + ((size_t)b * 1024 + kt * 32 + rv) * rs + h * 64 + c8 + 2048);
#else
    kreg = *(const v8bf*)(qkv + rowb + 1024);
    vreg = *(const v8bf*)(qkv + rowb + 2048);
#endif
  };
  auto stage = [&](int buf) {
#if HAS_TDM
    if (w == 0) __builtin_amdgcn_s_wait_tensorcnt((short)0);
#else
    *(v8bf*)&Ks[buf][rv * LDK + c8] = kreg;
#endif
#pragma unroll
    for (int i = 0; i < 8; ++i) Vs[buf][(c8 + i) * LDV + rv] = vreg[i];  // V^T
  };

  float mrow[8], lrow[8];
  v8f o[4];
#pragma unroll
  for (int r = 0; r < 8; ++r) { mrow[r] = -1e30f; lrow[r] = 0.0f; }
#pragma unroll
  for (int i = 0; i < 4; ++i) o[i] = {};

  bf16* ps = &Ps[w * 16 * LDP];

  fetch(0, 0);
  stage(0);
  __syncthreads();

  int cur = 0;
  for (int kt = 0; kt < 32; ++kt) {
    const int nxt = cur ^ 1;
    if (kt + 1 < 32) fetch(kt + 1, nxt);

    // S = Q.K^T : two 16x16 tiles over 32 key rows, head-dim via 2 chained WMMAs
    v8f s0 = {}, s1 = {};
#pragma unroll
    for (int c = 0; c < 2; ++c) {
      const v16bf k0 = frag_ld(&Ks[cur][(0 + lh) * LDK + c * 32 + kb]);
      const v16bf k1 = frag_ld(&Ks[cur][(16 + lh) * LDK + c * 32 + kb]);
      s0 = WMMA_BF16(qf[c], k0, s0);
      s1 = WMMA_BF16(qf[c], k1, s1);
    }

    // online softmax (row spans 16 lanes of each half-wave); scale pre-folded
#pragma unroll
    for (int r = 0; r < 8; ++r) {
      const float a0 = s0[r], a1 = s1[r];
      const float mx = red16_max(fmaxf(a0, a1));
      const float mnew = fmaxf(mrow[r], mx);
      const float corr = __expf(mrow[r] - mnew);
      const float p0 = __expf(a0 - mnew), p1 = __expf(a1 - mnew);
      const float sum = red16_sum(p0 + p1);
      lrow[r] = lrow[r] * corr + sum;
      mrow[r] = mnew;
#pragma unroll
      for (int i = 0; i < 4; ++i) o[i][r] *= corr;
      ps[(hh * 8 + r) * LDP + lh] = (bf16)p0;        // C-layout -> A-layout via LDS
      ps[(hh * 8 + r) * LDP + 16 + lh] = (bf16)p1;
    }

    // O += P.V
    const v16bf pf = frag_ld(&ps[lh * LDP + kb]);
#pragma unroll
    for (int i = 0; i < 4; ++i) {
      const v16bf vf = frag_ld(&Vs[cur][(i * 16 + lh) * LDV + kb]);
      o[i] = WMMA_BF16(pf, vf, o[i]);
    }

    if (kt + 1 < 32) stage(nxt);
    __syncthreads();
    cur = nxt;
  }

  // normalize, write bf16 (stays in L2 for the out-projection)
#pragma unroll
  for (int i = 0; i < 4; ++i) {
#pragma unroll
    for (int r = 0; r < 8; ++r) {
      const int row = qr0 + hh * 8 + r;
      const float val = o[i][r] / lrow[r];
      att[((size_t)b * 1024 + row) * 1024 + h * 64 + i * 16 + lh] = (bf16)val;
    }
  }
}

extern "C" void kernel_launch(void* const* d_in, const int* in_sizes, int n_in,
                              void* d_out, int out_size, void* d_ws, size_t ws_size,
                              hipStream_t stream) {
  const float* x = (const float*)d_in[0];      // [8,1024,1024]
  const float* w_qkv = (const float*)d_in[1];  // [1024,3072]
  const float* w_out = (const float*)d_in[2];  // [1024,1024]
  const float* b_out = (const float*)d_in[3];  // [1024]
  float* out = (float*)d_out;                  // [8,1024,1024]

  bf16* qkv_ws = (bf16*)d_ws;                   // 8192*3072 bf16
  bf16* att_ws = qkv_ws + (size_t)8192 * 3072;  // 8192*1024 bf16

  const dim3 blk(256);
  // Phase 1: qkv = bf16(x @ w_qkv), Q columns pre-scaled by dim^-0.5
  gemm128<3072, false, false, true, true>
      <<<dim3(24, 32), blk, 0, stream>>>(x, w_qkv, nullptr, qkv_ws, 1024);
  // Phase 2: flash attention per (b, h)
  attn_fa<<<dim3(8, 16, 8), blk, 0, stream>>>(qkv_ws, att_ws);
  // Phase 3: out = att @ w_out + b_out (fp32 output)
  gemm128<1024, true, true, false, false>
      <<<dim3(8, 32), blk, 0, stream>>>(att_ws, w_out, b_out, out, 1024);
}
